// Encoder_24283745092038
// MI455X (gfx1250) — matrix operand order repaired and checked by reading, not compiled, not run on hardware
//
#include <hip/hip_runtime.h>

// ---------------------------------------------------------------------------
// Types / WMMA helpers (gfx1250, wave32)
// ---------------------------------------------------------------------------
typedef __bf16 bf16;
typedef __attribute__((ext_vector_type(16))) __bf16 bf16x16;
typedef __attribute__((ext_vector_type(8)))  float  floatx8;
typedef __attribute__((ext_vector_type(4)))  unsigned int u32x4;
typedef __attribute__((ext_vector_type(8)))  unsigned int u32x8;

union FragU { bf16x16 v; uint4 u[2]; };

// A-matrix 16x32 bf16 fragment (interleaved K pattern per ISA table):
// lane<16: K = {b0..b0+7, b0+16..b0+23} with b0=0 ; lane>=16: b0=8
__device__ __forceinline__ bf16x16 load_fragA(const bf16* row, int laneHi) {
  FragU f;
  const bf16* p = row + (laneHi ? 8 : 0);
  f.u[0] = *reinterpret_cast<const uint4*>(p);
  f.u[1] = *reinterpret_cast<const uint4*>(p + 16);
  return f.v;
}
// B-matrix 32x16 bf16 fragment, sourced from BT[N][K] row (interleaved K, same as A):
// lane<16: K = {0..7, 16..23} ; lane>=16: K = {8..15, 24..31}
__device__ __forceinline__ bf16x16 load_fragB(const bf16* row, int laneHi) {
  FragU f;
  const bf16* p = row + (laneHi ? 8 : 0);
  f.u[0] = *reinterpret_cast<const uint4*>(p);
  f.u[1] = *reinterpret_cast<const uint4*>(p + 16);
  return f.v;
}
__device__ __forceinline__ floatx8 wmma_bf16(bf16x16 a, bf16x16 b, floatx8 c) {
  return __builtin_amdgcn_wmma_f32_16x16x32_bf16(false, a, false, b, (short)0, c,
                                                 false, false);
}

// ---------------------------------------------------------------------------
// CDNA5 async / TDM helpers
// ---------------------------------------------------------------------------
// LDS byte offset of a generic pointer into __shared__ memory (AS3 ptrs are
// 32-bit LDS offsets on amdgcn).
__device__ __forceinline__ unsigned lds_addr_of(const void* p) {
  return (unsigned)(unsigned long long)(__attribute__((address_space(3))) const char*)p;
}

// Per-lane async global->LDS 128-bit copy (tracked by ASYNCcnt).
__device__ __forceinline__ void async_load_b128(unsigned lds_byte_addr, const void* g) {
  asm volatile("global_load_async_to_lds_b128 %0, %1, off"
               :: "v"(lds_byte_addr), "v"((unsigned long long)(size_t)g)
               : "memory");
}
__device__ __forceinline__ void wait_asynccnt0() {
  asm volatile("s_wait_asynccnt 0x0" ::: "memory");
}

// Tensor Data Mover: 2D tile (tileH x tileW elements of 2 bytes) from a
// row-major [rows x K] bf16 tensor into LDS (contiguous). D# bitfields per
// CDNA5 ISA. Issued once per wave.
__device__ __forceinline__ void tdm_load_tile_2d(unsigned lds_byte_addr,
                                                 const void* tile_start,
                                                 unsigned K_elems, unsigned rows,
                                                 unsigned tileW, unsigned tileH) {
  unsigned long long ga = (unsigned long long)(size_t)tile_start;
  u32x4 g0;
  g0[0] = 1u;                                   // count=1, user mode
  g0[1] = lds_byte_addr;                        // lds_addr [63:32]
  g0[2] = (unsigned)ga;                         // global_addr [95:64]
  g0[3] = ((unsigned)(ga >> 32) & 0x01FFFFFFu)  // global_addr [120:96]
          | 0x80000000u;                        // type=2 ("image") [127:126]
  u32x8 g1;
  g1[0] = 1u << 16;                             // workgroup_mask=0, data_size=1 (2B)
  g1[1] = (K_elems & 0xFFFFu) << 16;            // tensor_dim0 lo16 -> [63:48]
  g1[2] = (K_elems >> 16) | ((rows & 0xFFFFu) << 16);   // dim0 hi / dim1 lo
  g1[3] = (rows >> 16) | (tileW << 16);         // dim1 hi / tile_dim0
  g1[4] = tileH;                                // tile_dim1 (tile_dim2=0)
  g1[5] = K_elems;                              // tensor_dim0_stride lo32
  g1[6] = 0u;                                   // stride0 hi / stride1 lo
  g1[7] = 0u;                                   // stride1 hi
  asm volatile("tensor_load_to_lds %0, %1" :: "s"(g0), "s"(g1) : "memory");
}

// ---------------------------------------------------------------------------
// Model constants
// ---------------------------------------------------------------------------
#define CD    1024      // model dim
#define CT    512       // seq len
#define CH    16        // heads
#define CHS   64        // head size
#define CB    16        // batch
#define CM    (CB*CT)   // 8192 rows
#define CDFF  4096
#define CL    4

// ---------------------------------------------------------------------------
// Embedding: h = tok_emb[x] + pos_emb
// ---------------------------------------------------------------------------
__global__ __launch_bounds__(256) void embed_kernel(
    const int* __restrict__ x, const float* __restrict__ tok,
    const float* __restrict__ pos, float* __restrict__ h) {
  size_t idx = (size_t)blockIdx.x * 256 + threadIdx.x;   // over M*D
  int d = (int)(idx & (CD - 1));
  int m = (int)(idx >> 10);
  int t = m & (CT - 1);
  int token = x[m];
  h[idx] = tok[(size_t)token * CD + d] + pos[(size_t)t * CD + d];
}

// ---------------------------------------------------------------------------
// LayerNorm (one block per row of 1024), writes bf16
// ---------------------------------------------------------------------------
__global__ __launch_bounds__(256) void ln_kernel(
    const float* __restrict__ h, const float* __restrict__ g,
    const float* __restrict__ b, bf16* __restrict__ z) {
  const int row = blockIdx.x;
  const float* x = h + (size_t)row * CD;
  float s = 0.f, s2 = 0.f;
  float vals[4];
#pragma unroll
  for (int i = 0; i < 4; ++i) {
    vals[i] = x[threadIdx.x + i * 256];
    s += vals[i]; s2 += vals[i] * vals[i];
  }
#pragma unroll
  for (int off = 16; off; off >>= 1) {
    s  += __shfl_xor(s,  off);
    s2 += __shfl_xor(s2, off);
  }
  __shared__ float ss[8], ss2[8];
  const int wave = threadIdx.x >> 5, lane = threadIdx.x & 31;
  if (lane == 0) { ss[wave] = s; ss2[wave] = s2; }
  __syncthreads();
  s = 0.f; s2 = 0.f;
#pragma unroll
  for (int i = 0; i < 8; ++i) { s += ss[i]; s2 += ss2[i]; }
  const float mean = s * (1.0f / CD);
  const float var  = s2 * (1.0f / CD) - mean * mean;
  const float inv  = rsqrtf(var + 1e-5f);
#pragma unroll
  for (int i = 0; i < 4; ++i) {
    int d = threadIdx.x + i * 256;
    float v = (vals[i] - mean) * inv * g[d] + b[d];
    z[(size_t)row * CD + d] = (bf16)v;
  }
}

// ---------------------------------------------------------------------------
// Weight prep: fp32 [K,N] -> bf16 [N,K]
// ---------------------------------------------------------------------------
__global__ __launch_bounds__(256) void transpose_cvt_kernel(
    const float* __restrict__ in, bf16* __restrict__ out, int K, int N) {
  size_t idx = (size_t)blockIdx.x * 256 + threadIdx.x;  // over K*N (in layout)
  int n = (int)(idx % N);
  size_t k = idx / N;
  out[(size_t)n * K + k] = (bf16)in[idx];
}

// Pack Wq/Wk/Wv (each [H,D,HS]) of one layer into bf16 [3072][1024] (row=N, col=K=D)
__global__ __launch_bounds__(256) void qkvT_kernel(
    const float* __restrict__ Wq, const float* __restrict__ Wk,
    const float* __restrict__ Wv, bf16* __restrict__ out) {
  int idx = blockIdx.x * 256 + threadIdx.x;  // over 3072*1024
  int d = idx & (CD - 1);
  int n = idx >> 10;                         // 0..3071
  int which = n >> 10;
  int r = n & (CD - 1);
  int hh = r >> 6, e = r & (CHS - 1);
  const float* W = (which == 0) ? Wq : (which == 1) ? Wk : Wv;
  out[(size_t)n * CD + d] = (bf16)W[((size_t)hh * CD + d) * CHS + e];
}

// ---------------------------------------------------------------------------
// Generic bf16 WMMA GEMM:  C[M,N] = A[M,K] * BT[N,K]^T   (BT is N-major)
// Block tile 128x64, 8 waves in 4x2, each wave 2x2 WMMA 16x16x32 tiles.
// A tile staged via TDM tensor_load_to_lds (wave 0), B tile via per-lane
// global_load_async_to_lds_b128 (ASYNCcnt).
// ---------------------------------------------------------------------------
#define MODE_QKV       0   // scatter bf16 to q,k (row-major) and vT (transposed)
#define MODE_RES_F32   1   // out0.f32 = acc + bias[n] + resid[m,n]
#define MODE_GELU_BF16 2   // out0.bf16 = gelu(acc + bias[n])

template <int MODE>
__global__ __launch_bounds__(256) void gemm_bf16_kernel(
    const bf16* __restrict__ A, const bf16* __restrict__ BT,
    const float* __restrict__ bias, const float* __restrict__ resid,
    void* __restrict__ out0, void* __restrict__ out1, void* __restrict__ out2,
    int M, int N, int K) {
  __shared__ __align__(16) bf16 As[128][32];
  __shared__ __align__(16) bf16 Bs[64][32];

  const int tid    = threadIdx.x;
  const int lane   = tid & 31;
  const int wave   = tid >> 5;
  const int wm     = wave >> 1;           // 0..3
  const int wn     = wave & 1;            // 0..1
  const int laneHi = (lane >> 4) & 1;
  const int ln15   = lane & 15;
  const int mBase  = blockIdx.y * 128;
  const int nBase  = blockIdx.x * 64;

  const int bRow = tid >> 2, bCol = (tid & 3) * 8;  // B-tile chunk for this lane
  const unsigned ldsA = lds_addr_of(&As[0][0]);
  const unsigned ldsB = lds_addr_of(&Bs[bRow][bCol]);

  floatx8 acc[2][2];
  const floatx8 zf = {0.f, 0.f, 0.f, 0.f, 0.f, 0.f, 0.f, 0.f};
#pragma unroll
  for (int i = 0; i < 2; ++i)
#pragma unroll
    for (int j = 0; j < 2; ++j) acc[i][j] = zf;

  for (int k0 = 0; k0 < K; k0 += 32) {
    // --- Stage A tile (128x32) with the Tensor Data Mover (one wave) ---
    if (wave == 0)
      tdm_load_tile_2d(ldsA, &A[(size_t)mBase * K + k0],
                       (unsigned)K, (unsigned)M, 32u, 128u);
    // --- Stage B tile (64x32) with async global->LDS b128 copies ---
    async_load_b128(ldsB, &BT[(size_t)(nBase + bRow) * K + k0 + bCol]);
    // Prefetch next K-step of B into cache
    if (k0 + 32 < K)
      __builtin_prefetch(&BT[(size_t)(nBase + bRow) * K + k0 + 32 + bCol], 0, 1);

    wait_asynccnt0();
    if (wave == 0) __builtin_amdgcn_s_wait_tensorcnt(0);
    __syncthreads();

    bf16x16 aF[2], bF[2];
    aF[0] = load_fragA(&As[wm * 32 + ln15][0], laneHi);
    aF[1] = load_fragA(&As[wm * 32 + 16 + ln15][0], laneHi);
    bF[0] = load_fragB(&Bs[wn * 32 + ln15][0], laneHi);
    bF[1] = load_fragB(&Bs[wn * 32 + 16 + ln15][0], laneHi);
#pragma unroll
    for (int i = 0; i < 2; ++i)
#pragma unroll
      for (int j = 0; j < 2; ++j)
        acc[i][j] = wmma_bf16(aF[i], bF[j], acc[i][j]);
    __syncthreads();
  }

  // Epilogue. C layout: elem r -> row (r + laneHi*8), col = ln15.
#pragma unroll
  for (int i = 0; i < 2; ++i)
#pragma unroll
    for (int j = 0; j < 2; ++j) {
      const int n = nBase + wn * 32 + j * 16 + ln15;
      const int mT = mBase + wm * 32 + i * 16 + laneHi * 8;
      if (MODE == MODE_QKV) {
        const int which = n >> 10;
        const int r2 = n & (CD - 1);
        const int hh = r2 >> 6, e = r2 & (CHS - 1);
#pragma unroll
        for (int r = 0; r < 8; ++r) {
          const int m = mT + r;
          const int b = m >> 9, t = m & (CT - 1);
          const float v = acc[i][j][r];
          if (which == 0)
            ((bf16*)out0)[(((size_t)(b * CH + hh) * CT + t) << 6) + e] = (bf16)v;
          else if (which == 1)
            ((bf16*)out1)[(((size_t)(b * CH + hh) * CT + t) << 6) + e] = (bf16)v;
          else
            ((bf16*)out2)[((size_t)(b * CH + hh) * CHS + e) * CT + t] = (bf16)v;
        }
      } else if (MODE == MODE_RES_F32) {
        const float bi = bias[n];
#pragma unroll
        for (int r = 0; r < 8; ++r) {
          const size_t idx = (size_t)(mT + r) * N + n;
          ((float*)out0)[idx] = acc[i][j][r] + bi + resid[idx];
        }
      } else {  // MODE_GELU_BF16
        const float bi = bias[n];
#pragma unroll
        for (int r = 0; r < 8; ++r) {
          const float xg = acc[i][j][r] + bi;
          const float gl = 0.5f * xg * (1.0f + erff(xg * 0.70710678118654752f));
          ((bf16*)out0)[(size_t)(mT + r) * N + n] = (bf16)gl;
        }
      }
    }
}

// ---------------------------------------------------------------------------
// Fused attention per (b,h, 16 query rows): S = qK^T*scale, softmax, O = P*V
// q,k : bf16 [B,H,T,HS]   vT : bf16 [B,H,HS,T]   O : bf16 [B,T,H*HS]
// ---------------------------------------------------------------------------
__global__ __launch_bounds__(256) void attn_kernel(
    const bf16* __restrict__ q, const bf16* __restrict__ k,
    const bf16* __restrict__ vT, bf16* __restrict__ O) {
  __shared__ __align__(16) float S[16][CT];          // 32 KB
  __shared__ __align__(16) bf16  P[16][CT];          // 16 KB
  __shared__ __align__(16) float Opart[2][16][CHS];  // 8 KB

  const int tid = threadIdx.x, lane = tid & 31, wave = tid >> 5;
  const int laneHi = (lane >> 4) & 1, ln15 = lane & 15;
  const int qt = blockIdx.x;         // 0..31 query tile
  const int bh = blockIdx.y;         // b*H + h
  const int b = bh >> 4, hh = bh & (CH - 1);

  const bf16* qbh = q  + (size_t)bh * CT * CHS;
  const bf16* kbh = k  + (size_t)bh * CT * CHS;
  const bf16* vbh = vT + (size_t)bh * CHS * CT;
  const floatx8 zf = {0.f, 0.f, 0.f, 0.f, 0.f, 0.f, 0.f, 0.f};

  // ---- Phase 1: S[16,512] = q(16x64) * k^T ; wave w covers 64 key columns
  bf16x16 aF0, aF1;
  {
    const bf16* qrow = qbh + (size_t)(qt * 16 + ln15) * CHS;
    aF0 = load_fragA(qrow, laneHi);
    aF1 = load_fragA(qrow + 32, laneHi);
  }
#pragma unroll
  for (int tn = 0; tn < 4; ++tn) {
    const int n0 = wave * 64 + tn * 16;
    const bf16* krow = kbh + (size_t)(n0 + ln15) * CHS;
    floatx8 accS = zf;
    accS = wmma_bf16(aF0, load_fragB(krow, laneHi), accS);
    accS = wmma_bf16(aF1, load_fragB(krow + 32, laneHi), accS);
#pragma unroll
    for (int r = 0; r < 8; ++r)
      S[r + laneHi * 8][n0 + ln15] = accS[r] * 0.125f;  // HS^-0.5
  }
  __syncthreads();

  // ---- Phase 2: softmax over each of 16 rows (2 rows per wave)
#pragma unroll
  for (int rr = 0; rr < 2; ++rr) {
    const int row = wave * 2 + rr;
    float mx = -3.4e38f;
    for (int c = lane; c < CT; c += 32) mx = fmaxf(mx, S[row][c]);
#pragma unroll
    for (int off = 16; off; off >>= 1) mx = fmaxf(mx, __shfl_xor(mx, off));
    float sum = 0.f;
    for (int c = lane; c < CT; c += 32) {
      float e = __expf(S[row][c] - mx);
      S[row][c] = e;
      sum += e;
    }
#pragma unroll
    for (int off = 16; off; off >>= 1) sum += __shfl_xor(sum, off);
    const float inv = 1.0f / sum;
    for (int c = lane; c < CT; c += 32) P[row][c] = (bf16)(S[row][c] * inv);
  }
  __syncthreads();

  // ---- Phase 3: O[16,64] = P[16,512] * V[512,64]; waves split (ntile, Khalf)
  {
    const int nt = wave & 3, kh = wave >> 2;
    const bf16* vrow = vbh + (size_t)(nt * 16 + ln15) * CT;
    floatx8 accO = zf;
#pragma unroll
    for (int ks = 0; ks < 8; ++ks) {
      const int kOff = kh * 256 + ks * 32;
      accO = wmma_bf16(load_fragA(&P[ln15][kOff], laneHi),
                       load_fragB(vrow + kOff, laneHi), accO);
    }
#pragma unroll
    for (int r = 0; r < 8; ++r)
      Opart[kh][r + laneHi * 8][nt * 16 + ln15] = accO[r];
  }
  __syncthreads();

  for (int i = tid; i < 16 * CHS; i += 256) {
    const int row = i >> 6, e = i & (CHS - 1);
    const float v = Opart[0][row][e] + Opart[1][row][e];
    const int t = qt * 16 + row;
    O[(((size_t)(b * CT + t)) * CH + hh) * CHS + e] = (bf16)v;
  }
}

// ---------------------------------------------------------------------------
// Head: logits[b,c] = sum_i hf[b,i] * Wout[i,c] + bout[c]   (NC = 3)
// ---------------------------------------------------------------------------
__global__ __launch_bounds__(256) void head_kernel(
    const bf16* __restrict__ hf, const float* __restrict__ Wout,
    const float* __restrict__ bout, float* __restrict__ out) {
  const int b = blockIdx.x;
  const bf16* hb = hf + (size_t)b * (CT * CD);
  float p0 = 0.f, p1 = 0.f, p2 = 0.f;
  for (int i = threadIdx.x; i < CT * CD; i += 256) {
    const float v = (float)hb[i];
    const float* w = Wout + (size_t)i * 3;
    p0 += v * w[0]; p1 += v * w[1]; p2 += v * w[2];
  }
#pragma unroll
  for (int off = 16; off; off >>= 1) {
    p0 += __shfl_xor(p0, off);
    p1 += __shfl_xor(p1, off);
    p2 += __shfl_xor(p2, off);
  }
  __shared__ float red[8][3];
  const int wave = threadIdx.x >> 5, lane = threadIdx.x & 31;
  if (lane == 0) { red[wave][0] = p0; red[wave][1] = p1; red[wave][2] = p2; }
  __syncthreads();
  if (threadIdx.x == 0) {
    float a0 = 0.f, a1 = 0.f, a2 = 0.f;
#pragma unroll
    for (int i = 0; i < 8; ++i) { a0 += red[i][0]; a1 += red[i][1]; a2 += red[i][2]; }
    out[b * 3 + 0] = a0 + bout[0];
    out[b * 3 + 1] = a1 + bout[1];
    out[b * 3 + 2] = a2 + bout[2];
  }
}

// ---------------------------------------------------------------------------
// Launcher
// ---------------------------------------------------------------------------
extern "C" void kernel_launch(void* const* d_in, const int* in_sizes, int n_in,
                              void* d_out, int out_size, void* d_ws, size_t ws_size,
                              hipStream_t stream) {
  (void)in_sizes; (void)n_in; (void)out_size; (void)ws_size;

  const int*   x       = (const int*)  d_in[0];
  const float* tok_emb = (const float*)d_in[1];
  const float* pos_emb = (const float*)d_in[2];
  const float* Wq      = (const float*)d_in[3];
  const float* Wk      = (const float*)d_in[4];
  const float* Wv      = (const float*)d_in[5];
  const float* Wproj   = (const float*)d_in[6];
  const float* bproj   = (const float*)d_in[7];
  const float* ln1_g   = (const float*)d_in[8];
  const float* ln1_b   = (const float*)d_in[9];
  const float* ln2_g   = (const float*)d_in[10];
  const float* ln2_b   = (const float*)d_in[11];
  const float* W1      = (const float*)d_in[12];
  const float* b1      = (const float*)d_in[13];
  const float* W2      = (const float*)d_in[14];
  const float* b2      = (const float*)d_in[15];
  const float* lnf_g   = (const float*)d_in[16];
  const float* lnf_b   = (const float*)d_in[17];
  const float* Wout    = (const float*)d_in[18];
  const float* bout    = (const float*)d_in[19];

  // Workspace carve-up
  char* ws = (char*)d_ws;
  size_t off = 0;
  auto carve = [&](size_t bytes) { void* p = ws + off; off += (bytes + 255) & ~(size_t)255; return p; };

  float* h      = (float*)carve((size_t)CM * CD * 4);            // 32 MB
  bf16*  z      = (bf16*) carve((size_t)CM * CD * 2);            // 16 MB
  bf16*  qb     = (bf16*) carve((size_t)CB * CH * CT * CHS * 2); // 16 MB
  bf16*  kb     = (bf16*) carve((size_t)CB * CH * CT * CHS * 2);
  bf16*  vTb    = (bf16*) carve((size_t)CB * CH * CHS * CT * 2);
  bf16*  Ob     = (bf16*) carve((size_t)CM * CD * 2);
  bf16*  a1     = (bf16*) carve((size_t)CM * CDFF * 2);          // 64 MB
  bf16*  hf     = (bf16*) carve((size_t)CM * CD * 2);
  bf16*  WqkvT  = (bf16*) carve((size_t)CL * 3 * CD * CD * 2);
  bf16*  WprojT = (bf16*) carve((size_t)CL * CD * CD * 2);
  bf16*  W1T    = (bf16*) carve((size_t)CL * CDFF * CD * 2);
  bf16*  W2T    = (bf16*) carve((size_t)CL * CD * CDFF * 2);

  // ---- Weight conversion (fp32 -> bf16, transposed to [N,K]) ----
  for (int l = 0; l < CL; ++l) {
    const size_t wOff = (size_t)l * CH * CD * CHS;
    qkvT_kernel<<<(3 * CD * CD) / 256, 256, 0, stream>>>(
        Wq + wOff, Wk + wOff, Wv + wOff, WqkvT + (size_t)l * 3 * CD * CD);
    transpose_cvt_kernel<<<(CD * CD) / 256, 256, 0, stream>>>(
        Wproj + (size_t)l * CD * CD, WprojT + (size_t)l * CD * CD, CD, CD);
    transpose_cvt_kernel<<<(CD * CDFF) / 256, 256, 0, stream>>>(
        W1 + (size_t)l * CD * CDFF, W1T + (size_t)l * CDFF * CD, CD, CDFF);
    transpose_cvt_kernel<<<(CDFF * CD) / 256, 256, 0, stream>>>(
        W2 + (size_t)l * CDFF * CD, W2T + (size_t)l * CD * CDFF, CDFF, CD);
  }

  // ---- Embedding ----
  embed_kernel<<<(size_t)CM * CD / 256, 256, 0, stream>>>(x, tok_emb, pos_emb, h);

  const dim3 gQKV(3 * CD / 64, CM / 128);
  const dim3 gProj(CD / 64, CM / 128);
  const dim3 gF1(CDFF / 64, CM / 128);
  const dim3 gF2(CD / 64, CM / 128);
  const dim3 gAtt(CT / 16, CB * CH);

  for (int l = 0; l < CL; ++l) {
    ln_kernel<<<CM, 256, 0, stream>>>(h, ln1_g + l * CD, ln1_b + l * CD, z);
    gemm_bf16_kernel<MODE_QKV><<<gQKV, 256, 0, stream>>>(
        z, WqkvT + (size_t)l * 3 * CD * CD, nullptr, nullptr,
        qb, kb, vTb, CM, 3 * CD, CD);
    attn_kernel<<<gAtt, 256, 0, stream>>>(qb, kb, vTb, Ob);
    gemm_bf16_kernel<MODE_RES_F32><<<gProj, 256, 0, stream>>>(
        Ob, WprojT + (size_t)l * CD * CD, bproj + l * CD, h,
        h, nullptr, nullptr, CM, CD, CD);
    ln_kernel<<<CM, 256, 0, stream>>>(h, ln2_g + l * CD, ln2_b + l * CD, z);
    gemm_bf16_kernel<MODE_GELU_BF16><<<gF1, 256, 0, stream>>>(
        z, W1T + (size_t)l * CDFF * CD, b1 + l * CDFF, nullptr,
        a1, nullptr, nullptr, CM, CDFF, CD);
    gemm_bf16_kernel<MODE_RES_F32><<<gF2, 256, 0, stream>>>(
        a1, W2T + (size_t)l * CD * CDFF, b2 + l * CD, h,
        h, nullptr, nullptr, CM, CD, CDFF);
  }

  ln_kernel<<<CM, 256, 0, stream>>>(h, lnf_g, lnf_b, hf);
  head_kernel<<<CB, 256, 0, stream>>>(hf, Wout, bout, (float*)d_out);
}
